// ImprovedAttention_76424648065937
// MI455X (gfx1250) — compile-verified, matrix operand order checked by reading, not run on hardware
//
#include <hip/hip_runtime.h>
#include <hip/hip_bf16.h>

// Problem constants (from reference)
#define B_   2
#define S_   2048
#define H_   16
#define D_   64
#define E_   1024
#define N3_  3072   // 3*E

typedef __attribute__((ext_vector_type(16))) __bf16 v16bf;
typedef __attribute__((ext_vector_type(8)))  float  v8f;

union Frag {
    v16bf  v;
    uint4  u[2];
    __bf16 e[16];
};

// ---- gfx1250 async global->LDS copy (ASYNCcnt path), per-lane 16B ----
__device__ __forceinline__ void async_copy_b128(unsigned lds_off, const void* gptr) {
    asm volatile("global_load_async_to_lds_b128 %0, %1, off"
                 :: "v"(lds_off), "v"((unsigned long long)(size_t)gptr)
                 : "memory");
}
__device__ __forceinline__ void wait_asynccnt0() {
    asm volatile("s_wait_asynccnt 0" ::: "memory");
}
__device__ __forceinline__ unsigned lds_offset(const void* p) {
    return (unsigned)(size_t)p;   // low 32 bits of generic LDS address = wave-relative LDS byte offset
}

// ---------------------------------------------------------------------------
// Kernel 1: QKV projection (bf16 WMMA, fp32 accum) + fused RoPE epilogue.
//   C[4096,3072] = X[4096,1024] * W[1024,3072] + b
//   Writes into one workspace: Q [B,H,S,D] | K [B,H,S,D] | V^T [B,H,D,S] (bf16).
// Block: 256 threads = 8 waves; block tile 64(M) x 128(N); wave tile 32x32.
// ---------------------------------------------------------------------------
__global__ __launch_bounds__(256)
void qkv_rope_kernel(const float* __restrict__ x,
                     const float* __restrict__ w,
                     const float* __restrict__ bias,
                     __bf16* __restrict__ wsb) {
    __shared__ __align__(16) __bf16 sX[64 * 32];    // [row][k] row-major
    __shared__ __align__(16) __bf16 sW[128 * 32];   // transposed: [n][k]

    const int tid  = threadIdx.x;
    const int wave = tid >> 5;
    const int lane = tid & 31;
    const int l16  = lane & 15;
    const int lh   = lane >> 4;          // half-wave select (0/1)
    const int wr   = wave >> 2;          // wave row group (0..1)
    const int wc   = wave & 3;           // wave col group (0..3)
    const int mblock = blockIdx.y * 64;
    const int nblock = blockIdx.x * 128;

    v8f acc[2][2];
    for (int i = 0; i < 2; ++i)
        for (int j = 0; j < 2; ++j)
            for (int r = 0; r < 8; ++r) acc[i][j][r] = 0.0f;

    const int xrow = tid >> 2;           // 0..63
    const int xk0  = (tid & 3) * 8;      // 0,8,16,24
    const int wk   = tid >> 3;           // 0..31
    const int wn0  = (tid & 7) * 16;     // 0..112

    for (int kt = 0; kt < E_ / 32; ++kt) {
        const int kbase = kt * 32;

        // ---- global loads (fp32) ----
        const float4* xp =
            reinterpret_cast<const float4*>(x + (size_t)(mblock + xrow) * E_ + kbase + xk0);
        float4 a0 = xp[0], a1 = xp[1];

        const float* wp = w + (size_t)(kbase + wk) * N3_ + nblock + wn0;
        float4 w0 = reinterpret_cast<const float4*>(wp)[0];
        float4 w1 = reinterpret_cast<const float4*>(wp)[1];
        float4 w2 = reinterpret_cast<const float4*>(wp)[2];
        float4 w3 = reinterpret_cast<const float4*>(wp)[3];

        __syncthreads();   // previous iteration's LDS reads done

        // ---- LDS stores (bf16) ----
        union { __bf16 e[8]; uint4 u; } xr;
        xr.e[0] = (__bf16)a0.x; xr.e[1] = (__bf16)a0.y;
        xr.e[2] = (__bf16)a0.z; xr.e[3] = (__bf16)a0.w;
        xr.e[4] = (__bf16)a1.x; xr.e[5] = (__bf16)a1.y;
        xr.e[6] = (__bf16)a1.z; xr.e[7] = (__bf16)a1.w;
        *reinterpret_cast<uint4*>(&sX[xrow * 32 + xk0]) = xr.u;

        float wv[16] = {w0.x, w0.y, w0.z, w0.w, w1.x, w1.y, w1.z, w1.w,
                        w2.x, w2.y, w2.z, w2.w, w3.x, w3.y, w3.z, w3.w};
        #pragma unroll
        for (int j = 0; j < 16; ++j)
            sW[(wn0 + j) * 32 + wk] = (__bf16)wv[j];

        __syncthreads();

        // ---- fragment loads + 4 WMMAs ----
        Frag afr[2], bfr[2];
        #pragma unroll
        for (int ti = 0; ti < 2; ++ti) {
            const int row = wr * 32 + ti * 16 + l16;
            const int off = lh * 8;
            afr[ti].u[0] = *reinterpret_cast<const uint4*>(&sX[row * 32 + off]);
            afr[ti].u[1] = *reinterpret_cast<const uint4*>(&sX[row * 32 + off + 16]);
        }
        #pragma unroll
        for (int tj = 0; tj < 2; ++tj) {
            const int col = wc * 32 + tj * 16 + l16;
            const int off = lh * 8;
            bfr[tj].u[0] = *reinterpret_cast<const uint4*>(&sW[col * 32 + off]);
            bfr[tj].u[1] = *reinterpret_cast<const uint4*>(&sW[col * 32 + off + 16]);
        }
        #pragma unroll
        for (int ti = 0; ti < 2; ++ti)
            #pragma unroll
            for (int tj = 0; tj < 2; ++tj)
                acc[ti][tj] = __builtin_amdgcn_wmma_f32_16x16x32_bf16(
                    false, afr[ti].v, false, bfr[tj].v,
                    (short)0, acc[ti][tj], false, false);
    }

    // ---- epilogue: bias + RoPE + branchless scatter to Q / K / V^T (bf16) ----
    const int  cbase = nblock + wc * 32;                 // 32-aligned
    const bool ropeg = ((cbase & 63) == 0) && (cbase < 2 * E_);  // first half of Q/K head
    const float bias0 = bias[cbase + l16];
    const float bias1 = bias[cbase + 16 + l16];
    const unsigned perbuf = (unsigned)B_ * H_ * S_ * D_;         // 4M elements

    #pragma unroll
    for (int ti = 0; ti < 2; ++ti) {
        const int mrow = mblock + wr * 32 + ti * 16 + lh * 8;  // rows mrow..mrow+7

        #pragma unroll
        for (int r = 0; r < 8; ++r) { acc[ti][0][r] += bias0; acc[ti][1][r] += bias1; }

        if (ropeg) {
            #pragma unroll
            for (int r = 0; r < 8; ++r) {
                const int   s  = (mrow + r) & (S_ - 1);
                const float a0 = (float)(s * 64 + l16);
                const float a1 = a0 + 16.0f;
                const float c0 = __cosf(a0), s0 = __sinf(a0);
                const float c1 = __cosf(a1), s1 = __sinf(a1);
                const float t0 = acc[ti][0][r], t1 = acc[ti][1][r];
                acc[ti][0][r] = t0 * c0 - t1 * s0;   // d in [0,16)
                acc[ti][1][r] = t1 * c1 + t0 * s1;   // d in [16,32)
            }
        }

        #pragma unroll
        for (int tj = 0; tj < 2; ++tj) {
            const unsigned col   = (unsigned)(cbase + tj * 16 + l16);
            const unsigned which = col >> 10;     // 0=Q 1=K 2=V
            const unsigned nh    = col & 1023;
            const unsigned h     = nh >> 6;
            const unsigned d     = nh & 63;
            const bool isv = (which == 2);
            // column-dependent part of the flat workspace index (branchless select)
            const unsigned colpart = isv
                ? (2u * perbuf + h * (unsigned)(D_ * S_) + d * (unsigned)S_)
                : (which * perbuf + h * (unsigned)(S_ * D_) + d);
            #pragma unroll
            for (int r = 0; r < 8; ++r) {
                const unsigned m = (unsigned)(mrow + r);
                const unsigned b = m >> 11;
                const unsigned s = m & (S_ - 1);
                const unsigned rowpart = b * (unsigned)(H_ * S_ * D_) + (isv ? s : s * (unsigned)D_);
                wsb[colpart + rowpart] = (__bf16)acc[ti][tj][r];
            }
        }
    }
}

// ---------------------------------------------------------------------------
// Kernel 2: causal flash attention, transposed formulation, with
// block-cooperative double-buffered async global->LDS staging of K and V^T.
//   Per wave: 16 query columns; block streams kv in tiles of 32.
//   S^T = K(16x32d) x Q^T(32d x 16q)  -> softmax stats are lane-local.
//   S^T C-regs ARE the B-fragment for O^T = V^T(16d x 32kv) x P^T(32kv x 16q).
// Block: 256 threads = 8 waves -> 128 query rows per block.
// ---------------------------------------------------------------------------
__global__ __launch_bounds__(256)
void flash_attn_kernel(const __bf16* __restrict__ qb,
                       const __bf16* __restrict__ kb,
                       const __bf16* __restrict__ vt,
                       float* __restrict__ out) {
    __shared__ __align__(16) __bf16 sK[2][32 * 64];   // [buf][kv][d]
    __shared__ __align__(16) __bf16 sV[2][64 * 32];   // [buf][d][kv]

    const int tid  = threadIdx.x;
    const int wave = tid >> 5;
    const int lane = tid & 31;
    const int l16  = lane & 15;
    const int lh   = lane >> 4;
    const int bh   = blockIdx.y;                 // b*H + h
    const int qbase = blockIdx.x * 128 + wave * 16;
    const int qrow  = qbase + l16;               // this lane's q column

    const __bf16* Q = qb + (size_t)bh * S_ * D_;
    const __bf16* K = kb + (size_t)bh * S_ * D_;
    const __bf16* V = vt + (size_t)bh * D_ * S_; // transposed [d][s]
    float*        O = out + (size_t)bh * S_ * D_;

    // staging assignment (per thread, one 16B async copy each for K and V^T)
    const int krow = tid >> 3, kchk = (tid & 7) * 8;   // K tile: [32][64]
    const int vrow = tid >> 2, vchk = (tid & 3) * 8;   // V tile: [64][32]
    const unsigned ldsK0 = lds_offset(&sK[0][krow * 64 + kchk]);
    const unsigned ldsK1 = lds_offset(&sK[1][krow * 64 + kchk]);
    const unsigned ldsV0 = lds_offset(&sV[0][vrow * 32 + vchk]);
    const unsigned ldsV1 = lds_offset(&sV[1][vrow * 32 + vchk]);

    // Q^T B-fragments, two 32-wide d chunks (resident for the whole kv loop)
    Frag qf[2];
    #pragma unroll
    for (int dc = 0; dc < 2; ++dc) {
        const __bf16* p = Q + (size_t)qrow * D_ + dc * 32 + lh * 8;
        qf[dc].u[0] = *reinterpret_cast<const uint4*>(p);
        qf[dc].u[1] = *reinterpret_cast<const uint4*>(p + 16);
    }

    v8f oacc[4];
    for (int dt = 0; dt < 4; ++dt)
        for (int r = 0; r < 8; ++r) oacc[dt][r] = 0.0f;
    float mrow = -__builtin_inff();
    float lrow = 0.0f;

    // block-uniform kv step count (largest causal extent in this block)
    const int nblk = blockIdx.x * 4 + 4;   // ((qbase_block+127)>>5)+1

    // prologue: stage tile 0 into buffer 0
    async_copy_b128(ldsK0, K + (size_t)krow * D_ + kchk);
    async_copy_b128(ldsV0, V + (size_t)vrow * S_ + vchk);

    for (int st = 0; st < nblk; ++st) {
        const int kv0 = st * 32;
        const int buf = st & 1;

        wait_asynccnt0();     // current tile landed in LDS (per-wave)
        __syncthreads();      // visible block-wide; prior buffer fully consumed

        // issue next tile into the other buffer (overlaps with compute below)
        if (st + 1 < nblk) {
            const int kvn = kv0 + 32;
            async_copy_b128(buf ? ldsK0 : ldsK1, K + (size_t)(kvn + krow) * D_ + kchk);
            async_copy_b128(buf ? ldsV0 : ldsV1, V + (size_t)vrow * S_ + kvn + vchk);
        }

        if (kv0 <= qbase + 15) {   // causal: this wave still has work in this tile
            // ---- S^T = K . Q^T : two 16(kv) x 16(q) C tiles ----
            v8f sc[2];
            for (int kvt = 0; kvt < 2; ++kvt)
                for (int r = 0; r < 8; ++r) sc[kvt][r] = 0.0f;

            #pragma unroll
            for (int kvt = 0; kvt < 2; ++kvt) {
                const int kr = kvt * 16 + l16;
                #pragma unroll
                for (int dc = 0; dc < 2; ++dc) {
                    Frag kf;
                    const __bf16* p = &sK[buf][kr * 64 + dc * 32 + lh * 8];
                    kf.u[0] = *reinterpret_cast<const uint4*>(p);
                    kf.u[1] = *reinterpret_cast<const uint4*>(p + 16);
                    sc[kvt] = __builtin_amdgcn_wmma_f32_16x16x32_bf16(
                        false, kf.v, false, qf[dc].v, (short)0, sc[kvt], false, false);
                }
            }

            // ---- scale + causal mask + running max ----
            float tmax = -__builtin_inff();
            #pragma unroll
            for (int kvt = 0; kvt < 2; ++kvt)
                #pragma unroll
                for (int r = 0; r < 8; ++r) {
                    const int kv = kv0 + kvt * 16 + lh * 8 + r;
                    float v = sc[kvt][r] * 0.125f;            // 1/sqrt(64)
                    v = (kv <= qrow) ? v : -__builtin_inff();
                    sc[kvt][r] = v;
                    tmax = fmaxf(tmax, v);
                }
            tmax = fmaxf(tmax, __shfl_xor(tmax, 16, 32));     // combine lane pair
            const float mnew  = fmaxf(mrow, tmax);
            const float scale = __expf(mrow - mnew);

            // ---- P^T = exp(S^T - m): C-regs become the PV B-fragment directly ----
            Frag pf;
            float tsum = 0.0f;
            #pragma unroll
            for (int kvt = 0; kvt < 2; ++kvt)
                #pragma unroll
                for (int r = 0; r < 8; ++r) {
                    const float p = __expf(sc[kvt][r] - mnew);
                    tsum += p;
                    pf.e[kvt * 8 + r] = (__bf16)p;
                }
            tsum += __shfl_xor(tsum, 16, 32);
            lrow = lrow * scale + tsum;
            mrow = mnew;

            // ---- O^T += V^T . P^T : four 16(d) x 16(q) tiles ----
            #pragma unroll
            for (int dt = 0; dt < 4; ++dt) {
                #pragma unroll
                for (int r = 0; r < 8; ++r) oacc[dt][r] *= scale;
                Frag vf;
                const __bf16* p = &sV[buf][(dt * 16 + l16) * 32 + lh * 8];
                vf.u[0] = *reinterpret_cast<const uint4*>(p);
                vf.u[1] = *reinterpret_cast<const uint4*>(p + 16);
                oacc[dt] = __builtin_amdgcn_wmma_f32_16x16x32_bf16(
                    false, vf.v, false, pf.v, (short)0, oacc[dt], false, false);
            }
        }
    }

    // ---- normalize and store fp32 output in bhqd layout ----
    const float inv = 1.0f / lrow;
    #pragma unroll
    for (int dt = 0; dt < 4; ++dt) {
        float4 o0, o1;
        o0.x = oacc[dt][0] * inv; o0.y = oacc[dt][1] * inv;
        o0.z = oacc[dt][2] * inv; o0.w = oacc[dt][3] * inv;
        o1.x = oacc[dt][4] * inv; o1.y = oacc[dt][5] * inv;
        o1.z = oacc[dt][6] * inv; o1.w = oacc[dt][7] * inv;
        float* p = O + (size_t)qrow * D_ + dt * 16 + lh * 8;
        *reinterpret_cast<float4*>(p)     = o0;
        *reinterpret_cast<float4*>(p + 4) = o1;
    }
}

// ---------------------------------------------------------------------------
extern "C" void kernel_launch(void* const* d_in, const int* in_sizes, int n_in,
                              void* d_out, int out_size, void* d_ws, size_t ws_size,
                              hipStream_t stream) {
    const float* x    = (const float*)d_in[0];
    // d_in[1] is the causal mask: implicit in the kernel, unused.
    const float* W    = (const float*)d_in[2];
    const float* bias = (const float*)d_in[3];

    const size_t perbuf = (size_t)B_ * H_ * S_ * D_;   // 4M bf16 = 8 MB each
    __bf16* wsb = (__bf16*)d_ws;                       // Q | K | V^T
    __bf16* qb = wsb;
    __bf16* kb = wsb + perbuf;
    __bf16* vt = wsb + 2 * perbuf;

    dim3 g1(N3_ / 128, (B_ * S_) / 64);
    qkv_rope_kernel<<<g1, 256, 0, stream>>>(x, W, bias, wsb);

    dim3 g2(S_ / 128, B_ * H_);
    flash_attn_kernel<<<g2, 256, 0, stream>>>(qb, kb, vt, (float*)d_out);
}
